// Simple_DePatch_28664611734071
// MI455X (gfx1250) — matrix operand order repaired and checked by reading, not compiled, not run on hardware
//
#include <hip/hip_runtime.h>
#include <hip/hip_bf16.h>
#include <math.h>

typedef _Float16 v16h __attribute__((ext_vector_type(16)));
typedef _Float16 v8h  __attribute__((ext_vector_type(8)));
typedef float    v8f  __attribute__((ext_vector_type(8)));

#define Bq   128
#define CIN  3
#define Hh   224
#define Ww   224
#define Pc   14
#define Np   196            // 14*14 patches
#define Dd   384
#define Kk   768            // CIN*16*16
#define MT   7              // 7 M-tiles of 32 rows per batch (224 rows, 196 valid)

// workspace byte offsets
#define O_BT      0          // f16 Bt[384][768]            = 589824 B
#define O_BETA    589824     // f32 beta[384]
#define O_RHO     591360     // f32 rho[384][3]
#define O_STATS   595968     // f32 stats[13]: Vb, Mb[3], M2[3][3]
#define O_CENTERS 596224     // f32 centers[25088][2]

// ---------------------------------------------------------------------------
// prep: Bt = (f16)conv_w  (layout already matches k = c*256+py*16+px);
// rw[d][c] = sum_p out_w[d][3p+c]; beta/rho = centered out_b / rw;
// stats: Vb = mean(beta^2), Mb[c] = mean(beta*rho_c), M2[c][c'] = mean(rho_c*rho_c')
// ---------------------------------------------------------------------------
__global__ __launch_bounds__(256) void prep_kernel(
    const float* __restrict__ conv_w, const float* __restrict__ out_w,
    const float* __restrict__ out_b, char* __restrict__ ws)
{
    _Float16* Bt   = (_Float16*)(ws + O_BT);
    float*    beta = (float*)(ws + O_BETA);
    float*    rho  = (float*)(ws + O_RHO);
    float*    stats= (float*)(ws + O_STATS);

    __shared__ float rwS[Dd * 3];
    __shared__ float bS[Dd];
    __shared__ float meanS[4];

    const int tid = threadIdx.x;

    for (int i = tid; i < Dd * Kk; i += 256)
        Bt[i] = (_Float16)conv_w[i];

    for (int d = tid; d < Dd; d += 256) {
        const float* wr = out_w + (size_t)d * Kk;
        float s0 = 0.f, s1 = 0.f, s2 = 0.f;
        for (int p = 0; p < 256; ++p) {
            s0 += wr[3 * p + 0];
            s1 += wr[3 * p + 1];
            s2 += wr[3 * p + 2];
        }
        rwS[d * 3 + 0] = s0; rwS[d * 3 + 1] = s1; rwS[d * 3 + 2] = s2;
        bS[d] = out_b[d];
    }
    __syncthreads();

    if (tid == 0) {
        float mb = 0.f, m0 = 0.f, m1 = 0.f, m2 = 0.f;
        for (int d = 0; d < Dd; ++d) {
            mb += bS[d];
            m0 += rwS[d * 3 + 0]; m1 += rwS[d * 3 + 1]; m2 += rwS[d * 3 + 2];
        }
        meanS[0] = mb / Dd; meanS[1] = m0 / Dd; meanS[2] = m1 / Dd; meanS[3] = m2 / Dd;
    }
    __syncthreads();

    for (int d = tid; d < Dd; d += 256) {
        float be = bS[d] - meanS[0];
        beta[d] = be; bS[d] = be;
        for (int c = 0; c < 3; ++c) {
            float r = rwS[d * 3 + c] - meanS[1 + c];
            rho[d * 3 + c] = r; rwS[d * 3 + c] = r;
        }
    }
    __syncthreads();

    if (tid == 0) {
        float Vb = 0.f, Mb[3] = {0.f, 0.f, 0.f};
        float M2[9] = {0.f, 0.f, 0.f, 0.f, 0.f, 0.f, 0.f, 0.f, 0.f};
        for (int d = 0; d < Dd; ++d) {
            float be = bS[d];
            Vb += be * be;
            for (int c = 0; c < 3; ++c) {
                float rc = rwS[d * 3 + c];
                Mb[c] += be * rc;
                for (int c2 = 0; c2 < 3; ++c2)
                    M2[c * 3 + c2] += rc * rwS[d * 3 + c2];
            }
        }
        stats[0] = Vb / Dd;
        for (int c = 0; c < 3; ++c) stats[1 + c] = Mb[c] / Dd;
        for (int i = 0; i < 9; ++i) stats[4 + i] = M2[i] / Dd;
    }
}

// ---------------------------------------------------------------------------
// fused patch-embed WMMA GEMM + LN1 + GELU + offset head + tanh -> centers
// block = 256 threads (8 waves). Each block: one batch b, one 32-patch tile.
// Wave w: 2 M-subtiles x 3 N-tiles (n0 = 48w + {0,16,32}) -> 6 wmma per K step,
// B fragments loaded once per K step and reused across both M subtiles.
// ---------------------------------------------------------------------------
__device__ inline v8h load_cvt8(const float* __restrict__ p)
{
    const float4* q = (const float4*)p;
    float4 f0 = q[0], f1 = q[1];
    v8h r;
    r[0] = (_Float16)f0.x; r[1] = (_Float16)f0.y; r[2] = (_Float16)f0.z; r[3] = (_Float16)f0.w;
    r[4] = (_Float16)f1.x; r[5] = (_Float16)f1.y; r[6] = (_Float16)f1.z; r[7] = (_Float16)f1.w;
    return r;
}

__device__ inline v16h load_a_frag(const float* __restrict__ xb, bool valid,
                                   int gy, int gx, int k0, int hi)
{
    v8h alo = {0,0,0,0,0,0,0,0};
    v8h ahi = {0,0,0,0,0,0,0,0};
    if (valid) {
        int kk = k0 + hi * 16;
        {
            int c = kk >> 8, py = (kk >> 4) & 15, px = kk & 15;
            alo = load_cvt8(xb + ((size_t)(c * Hh + gy * 16 + py) * Ww + gx * 16 + px));
        }
        kk += 8;
        {
            int c = kk >> 8, py = (kk >> 4) & 15, px = kk & 15;
            ahi = load_cvt8(xb + ((size_t)(c * Hh + gy * 16 + py) * Ww + gx * 16 + px));
        }
    }
    return __builtin_shufflevector(alo, ahi,
                                   0,1,2,3,4,5,6,7,8,9,10,11,12,13,14,15);
}

__global__ __launch_bounds__(256) void embed_off_kernel(
    const float* __restrict__ x, const float* __restrict__ conv_b,
    const float* __restrict__ ln1_g, const float* __restrict__ ln1_b,
    const float* __restrict__ off_w, char* __restrict__ ws)
{
    const _Float16* Bt = (const _Float16*)(ws + O_BT);
    float* centers = (float*)(ws + O_CENTERS);

    __shared__ float tokLDS[32 * Dd];   // 48 KB

    const int tid  = threadIdx.x;
    const int wave = tid >> 5;
    const int lane = tid & 31;
    const int l16  = lane & 15;
    const int hi   = lane >> 4;

    const int b    = blockIdx.x / MT;
    const int tile = blockIdx.x % MT;

    // A-fragment rows (patches) this lane supplies for the two M subtiles
    const int  nr0 = tile * 32 + l16;
    const int  nr1 = nr0 + 16;
    const bool va0 = (nr0 < Np);
    const bool va1 = (nr1 < Np);
    const int  gy0 = nr0 / Pc, gx0 = nr0 % Pc;
    const int  gy1 = nr1 / Pc, gx1 = nr1 % Pc;
    const float* xb = x + (size_t)b * (CIN * Hh * Ww);

    v8f acc00 = {0.f,0.f,0.f,0.f,0.f,0.f,0.f,0.f};
    v8f acc01 = {0.f,0.f,0.f,0.f,0.f,0.f,0.f,0.f};
    v8f acc02 = {0.f,0.f,0.f,0.f,0.f,0.f,0.f,0.f};
    v8f acc10 = {0.f,0.f,0.f,0.f,0.f,0.f,0.f,0.f};
    v8f acc11 = {0.f,0.f,0.f,0.f,0.f,0.f,0.f,0.f};
    v8f acc12 = {0.f,0.f,0.f,0.f,0.f,0.f,0.f,0.f};

    const int n0 = wave * 48;
    // B layout used here (contiguous k, same order as A): lanes 0-15 col=l, K = k0..k0+15 contiguous;
    // lanes 16-31 col=l-16, K = k0+16..k0+31  -> one 32B v16h load per lane.
    const _Float16* Bl0 = Bt + (size_t)(n0 +  0 + l16) * Kk + hi * 16;
    const _Float16* Bl1 = Bt + (size_t)(n0 + 16 + l16) * Kk + hi * 16;
    const _Float16* Bl2 = Bt + (size_t)(n0 + 32 + l16) * Kk + hi * 16;

#pragma unroll 2
    for (int ks = 0; ks < 24; ++ks) {
        const int k0 = ks * 32;

        // prefetch next K step's B rows (global_prefetch_b8)
        if (ks < 23) {
            __builtin_prefetch(Bl0 + k0 + 32, 0, 1);
            __builtin_prefetch(Bl1 + k0 + 32, 0, 1);
            __builtin_prefetch(Bl2 + k0 + 32, 0, 1);
        }

        v16h b0 = *(const v16h*)(Bl0 + k0);
        v16h b1 = *(const v16h*)(Bl1 + k0);
        v16h b2 = *(const v16h*)(Bl2 + k0);

        v16h a0 = load_a_frag(xb, va0, gy0, gx0, k0, hi);
        v16h a1 = load_a_frag(xb, va1, gy1, gx1, k0, hi);

        acc00 = __builtin_amdgcn_wmma_f32_16x16x32_f16(false, a0, false, b0, (short)0, acc00, false, false);
        acc01 = __builtin_amdgcn_wmma_f32_16x16x32_f16(false, a0, false, b1, (short)0, acc01, false, false);
        acc02 = __builtin_amdgcn_wmma_f32_16x16x32_f16(false, a0, false, b2, (short)0, acc02, false, false);
        acc10 = __builtin_amdgcn_wmma_f32_16x16x32_f16(false, a1, false, b0, (short)0, acc10, false, false);
        acc11 = __builtin_amdgcn_wmma_f32_16x16x32_f16(false, a1, false, b1, (short)0, acc11, false, false);
        acc12 = __builtin_amdgcn_wmma_f32_16x16x32_f16(false, a1, false, b2, (short)0, acc12, false, false);
    }

    // C/D layout: VGPR v -> row M = v + 8*hi, col N = n0 + l16. Add conv bias.
    {
        float cb0 = conv_b[n0 +  0 + l16];
        float cb1 = conv_b[n0 + 16 + l16];
        float cb2 = conv_b[n0 + 32 + l16];
#pragma unroll
        for (int v = 0; v < 8; ++v) {
            int r0 = v + hi * 8;
            int r1 = r0 + 16;
            tokLDS[r0 * Dd + n0 +  0 + l16] = acc00[v] + cb0;
            tokLDS[r0 * Dd + n0 + 16 + l16] = acc01[v] + cb1;
            tokLDS[r0 * Dd + n0 + 32 + l16] = acc02[v] + cb2;
            tokLDS[r1 * Dd + n0 +  0 + l16] = acc10[v] + cb0;
            tokLDS[r1 * Dd + n0 + 16 + l16] = acc11[v] + cb1;
            tokLDS[r1 * Dd + n0 + 32 + l16] = acc12[v] + cb2;
        }
    }
    __syncthreads();

    // LN1 + GELU(exact) + 2-wide offset head + tanh -> centers.
    // 32 groups of 8 lanes, one row each; reductions via shfl_xor(width=8).
    {
        const int g  = tid >> 3;       // row 0..31
        const int sl = tid & 7;
        float s1 = 0.f, s2 = 0.f;
        for (int d = sl; d < Dd; d += 8) {
            float t = tokLDS[g * Dd + d];
            s1 += t; s2 += t * t;
        }
        for (int o = 4; o; o >>= 1) {
            s1 += __shfl_xor(s1, o, 8);
            s2 += __shfl_xor(s2, o, 8);
        }
        float mean = s1 / (float)Dd;
        float var  = s2 / (float)Dd - mean * mean;
        float inv  = rsqrtf(var + 1e-5f);

        float o0 = 0.f, o1 = 0.f;
        for (int d = sl; d < Dd; d += 8) {
            float t  = (tokLDS[g * Dd + d] - mean) * inv * ln1_g[d] + ln1_b[d];
            float ge = 0.5f * t * (1.0f + erff(t * 0.70710678118654752f));
            o0 += ge * off_w[d];
            o1 += ge * off_w[Dd + d];
        }
        for (int o = 4; o; o >>= 1) {
            o0 += __shfl_xor(o0, o, 8);
            o1 += __shfl_xor(o1, o, 8);
        }

        const int n = tile * 32 + g;
        if (sl == 0 && n < Np) {
            const float S = 1.0f / (float)Pc;
            int gy2 = n / Pc, gx2 = n % Pc;
            // centers == ctr: the wh terms cancel in (box_lo+box_hi)/2
            float cx = ((float)gx2 + 0.5f) * S + tanhf(o0) * S;
            float cy = ((float)gy2 + 0.5f) * S + tanhf(o1) * S;
            size_t row = (size_t)b * Np + n;
            centers[row * 2 + 0] = cx;
            centers[row * 2 + 1] = cy;
        }
    }
}

// ---------------------------------------------------------------------------
// bilinear sample at centers + rank-3 output GEMM + closed-form LN2
// block = 256 threads handles 16 rows.
// ---------------------------------------------------------------------------
__global__ __launch_bounds__(256) void sample_out_kernel(
    const float* __restrict__ x, const float* __restrict__ ln2_g,
    const float* __restrict__ ln2_b, const char* __restrict__ ws,
    float* __restrict__ out)
{
    const float* beta    = (const float*)(ws + O_BETA);
    const float* rho     = (const float*)(ws + O_RHO);
    const float* stats   = (const float*)(ws + O_STATS);
    const float* centers = (const float*)(ws + O_CENTERS);

    __shared__ float sS[16][3];
    __shared__ float invS[16];

    const int tid  = threadIdx.x;
    const int row0 = blockIdx.x * 16;

    if (tid < 16) {
        const int row = row0 + tid;
        const int b   = row / Np;
        float cx = centers[(size_t)row * 2 + 0];
        float cy = centers[(size_t)row * 2 + 1];
        float ix = cx * (float)Ww - 0.5f;
        float iy = cy * (float)Hh - 0.5f;
        float fx = floorf(ix), fy = floorf(iy);
        float wx = ix - fx,  wy = iy - fy;
        int x0 = (int)fx, y0 = (int)fy;
        int x1 = x0 + 1,  y1 = y0 + 1;
        bool vx0 = (x0 >= 0) & (x0 < Ww), vx1 = (x1 >= 0) & (x1 < Ww);
        bool vy0 = (y0 >= 0) & (y0 < Hh), vy1 = (y1 >= 0) & (y1 < Hh);
        float w00 = (1.f - wx) * (1.f - wy), w10 = wx * (1.f - wy);
        float w01 = (1.f - wx) * wy,         w11 = wx * wy;

        const float* xb = x + (size_t)b * (CIN * Hh * Ww);
        float s[3];
#pragma unroll
        for (int c = 0; c < 3; ++c) {
            const float* pc = xb + (size_t)c * Hh * Ww;
            float acc = 0.f;
            if (vy0 & vx0) acc += w00 * pc[y0 * Ww + x0];
            if (vy0 & vx1) acc += w10 * pc[y0 * Ww + x1];
            if (vy1 & vx0) acc += w01 * pc[y1 * Ww + x0];
            if (vy1 & vx1) acc += w11 * pc[y1 * Ww + x1];
            s[c] = acc;
            sS[tid][c] = acc;
        }
        // var of (out_b + rw @ s) over d, in closed form
        float var = stats[0];
#pragma unroll
        for (int c = 0; c < 3; ++c) var += 2.f * stats[1 + c] * s[c];
#pragma unroll
        for (int c = 0; c < 3; ++c)
#pragma unroll
            for (int c2 = 0; c2 < 3; ++c2)
                var += stats[4 + c * 3 + c2] * s[c] * s[c2];
        invS[tid] = rsqrtf(var + 1e-5f);
    }
    __syncthreads();

    for (int i = tid; i < 16 * Dd; i += 256) {
        int r = i / Dd;
        int d = i - r * Dd;
        float v = beta[d] + rho[d * 3 + 0] * sS[r][0]
                          + rho[d * 3 + 1] * sS[r][1]
                          + rho[d * 3 + 2] * sS[r][2];
        out[(size_t)(row0 + r) * Dd + d] = v * invS[r] * ln2_g[d] + ln2_b[d];
    }
}

// ---------------------------------------------------------------------------
extern "C" void kernel_launch(void* const* d_in, const int* in_sizes, int n_in,
                              void* d_out, int out_size, void* d_ws, size_t ws_size,
                              hipStream_t stream)
{
    const float* x      = (const float*)d_in[0];
    const float* conv_w = (const float*)d_in[1];
    const float* conv_b = (const float*)d_in[2];
    const float* ln1_g  = (const float*)d_in[3];
    const float* ln1_b  = (const float*)d_in[4];
    const float* off_w  = (const float*)d_in[5];
    const float* out_w  = (const float*)d_in[6];
    const float* out_b  = (const float*)d_in[7];
    const float* ln2_g  = (const float*)d_in[8];
    const float* ln2_b  = (const float*)d_in[9];

    char*  ws  = (char*)d_ws;
    float* out = (float*)d_out;

    prep_kernel<<<1, 256, 0, stream>>>(conv_w, out_w, out_b, ws);
    embed_off_kernel<<<Bq * MT, 256, 0, stream>>>(x, conv_b, ln1_g, ln1_b, off_w, ws);
    sample_out_kernel<<<(Bq * Np) / 16, 256, 0, stream>>>(x, ln2_g, ln2_b, ws, out);
}
